// LinearAttention_3109556322332
// MI455X (gfx1250) — compile-verified, matrix operand order checked by reading, not run on hardware
//
#include <hip/hip_runtime.h>

// ---------------------------------------------------------------------------
// LinearAttention (3D volume) fused for MI455X / gfx1250.
//  prep    : transpose weights to bf16, zero accumulators
//  phaseA  : kv = x@Wkv (WMMA bf16), E=exp(k), context += E·V^T (WMMA), Z += ΣE;
//            emits flat bf16 tile image of x via global_store_async_from_lds_b128
//  mid     : M[b] = (context/Z) @ W_out  (folds out-projection, 128x128/batch)
//  phaseB  : stages bf16 tile with TDM tensor_load_to_lds (s_wait_tensorcnt),
//            falling back to global_load_async_to_lds_b128; q = x@Wq (WMMA)
//            -> head softmax -> y = s@M (WMMA) -> +b_out -> LayerNorm
// ---------------------------------------------------------------------------

typedef __attribute__((ext_vector_type(16))) __bf16 v16bf;
typedef __attribute__((ext_vector_type(8)))  float  v8f;
typedef __attribute__((ext_vector_type(4)))  unsigned int v4u;
typedef __attribute__((ext_vector_type(8)))  int v8i;
typedef __attribute__((ext_vector_type(4)))  int v4i;
typedef unsigned short u16_t;

#define XS 132          // f32 LDS row stride (dword stride %64 = 4 -> conflict free)
#define ES 136          // bf16 LDS row stride (dword stride 68 %64 = 4, rows 16B aligned)
#define TILE_ELEMS (128 * ES)   // 17408 bf16 elements = one flat padded tile (34816 B)

static constexpr int SMEM_A = 3 * 128 * ES * 2;                   // Xbf,E,V = 104448 B
static constexpr int SMEM_B = 128 * ES * 2 + 128 * XS * 4 + 1024; // Xbf/S, Q/Y, stats

#if __has_builtin(__builtin_amdgcn_tensor_load_to_lds)
#define HAS_TDM 1
#else
#define HAS_TDM 0
#endif

__device__ __forceinline__ u16_t f2bf(float f) {   // native RNE f32->bf16
  union { __bf16 b; u16_t s; } c; c.b = (__bf16)f; return c.s;
}
__device__ __forceinline__ float bf2f(u16_t s) {
  union { u16_t s; __bf16 b; } c; c.s = s; return (float)c.b;
}
__device__ __forceinline__ uint2 cvt4(float4 v) {  // 4 x f32 -> packed 4 x bf16
  union { uint2 u; u16_t s[4]; } p;
  p.s[0] = f2bf(v.x); p.s[1] = f2bf(v.y); p.s[2] = f2bf(v.z); p.s[3] = f2bf(v.w);
  return p.u;
}

union FragU { v16bf v; uint4 u[2]; u16_t s[16]; };

__device__ __forceinline__ v8f wmma_bf16(v16bf a, v16bf b, v8f c) {
  return __builtin_amdgcn_wmma_f32_16x16x32_bf16(false, a, false, b, (short)0, c,
                                                 false, false);
}

// bf16 fragment (A rows / mirrored B columns): lanes 0-15 hold row rowBase+lane,
// K in {0..7,16..23}; lanes 16-31 K in {8..15,24..31}. Two b128 loads.
__device__ __forceinline__ v16bf frag_bf16(const u16_t* buf, int rowBase, int stride,
                                           int ks, int lane) {
  int row = rowBase + (lane & 15);
  int klo = ks * 32 + ((lane >> 4) & 1) * 8;
  const u16_t* p = buf + row * stride + klo;
  FragU f;
  f.u[0] = *(const uint4*)(p);
  f.u[1] = *(const uint4*)(p + 16);
  return f.v;
}

// ---- gfx1250 async / tensor data-mover helpers -----------------------------
__device__ __forceinline__ void wait_async0() {
#if __has_builtin(__builtin_amdgcn_s_wait_asynccnt)
  __builtin_amdgcn_s_wait_asynccnt(0);
#else
  asm volatile("s_wait_asynccnt 0" ::: "memory");
#endif
}
__device__ __forceinline__ void wait_tensor0() {
#if __has_builtin(__builtin_amdgcn_s_wait_tensorcnt)
  __builtin_amdgcn_s_wait_tensorcnt(0);
#else
  asm volatile("s_wait_tensorcnt 0x0" ::: "memory");
#endif
}
__device__ __forceinline__ void async_load_b128(unsigned lds_off, const void* gaddr) {
  asm volatile("global_load_async_to_lds_b128 %0, %1, off"
               :: "v"(lds_off), "v"(gaddr) : "memory");
}
__device__ __forceinline__ void async_store_b128(const void* gaddr, unsigned lds_off) {
  asm volatile("global_store_async_from_lds_b128 %0, %1, off"
               :: "v"(gaddr), "v"(lds_off) : "memory");
}
__device__ __forceinline__ unsigned lds_off_of(const void* p) {
  return (unsigned)(size_t)p;   // generic LDS pointer: low 32 bits = LDS offset
}

#if HAS_TDM
// D# for a flat 1-D copy of one padded tile (17408 bf16 = 34816 B), per ISA §8.
__device__ __forceinline__ void tdm_desc(v4u& g0, v8i& g1, unsigned lds,
                                         const void* gptr) {
  unsigned long long ga = (unsigned long long)(size_t)gptr;
  g0[0] = 1u;                                         // count=1, user descriptor
  g0[1] = lds;                                        // lds_addr (bytes)
  g0[2] = (unsigned)ga;                               // global_addr[31:0]
  g0[3] = (unsigned)((ga >> 32) & 0x01ffffffu) | (2u << 30);  // ga[56:32], type=2
  g1[0] = (int)(1u << 16);              // workgroup_mask=0, data_size=1 (2 bytes)
  g1[1] = (int)((TILE_ELEMS & 0xffffu) << 16);        // tensor_dim0[15:0]
  g1[2] = (int)((TILE_ELEMS >> 16) | (1u << 16));     // tensor_dim0 hi, tensor_dim1=1
  g1[3] = (int)((TILE_ELEMS & 0xffffu) << 16);        // tensor_dim1 hi, tile_dim0
  g1[4] = 0;                                          // tile_dim1/2 unused
  g1[5] = (int)TILE_ELEMS;                            // tensor_dim0_stride lo
  g1[6] = 0;
  g1[7] = 0;
}
// 6-arg form on this toolchain: (v4u g0, v8i g1, v4i, v4i, v8i, i32 cpol)
__device__ __forceinline__ void tdm_load(v4u g0, v8i g1) {
  v4i z4 = {};
  v8i z8 = {};
  __builtin_amdgcn_tensor_load_to_lds(g0, g1, z4, z4, z8, 0);
}
#endif

// ---------------------------------------------------------------------------
// prep: Wq^T (128x128) and Wkv^T (256x128) in bf16; zero ctx/Z accumulators.
// ---------------------------------------------------------------------------
__global__ void __launch_bounds__(256)
prep_kernel(const float* __restrict__ w_qkv, u16_t* __restrict__ WqT,
            u16_t* __restrict__ WkvT, float* __restrict__ g_ctx,
            float* __restrict__ g_zk) {
  int i = blockIdx.x * 256 + threadIdx.x;
  if (i < 128 * 384) {
    int cin = i / 384, dch = i % 384;
    u16_t v = f2bf(w_qkv[i]);
    if (dch < 128) WqT[dch * 128 + cin] = v;
    else           WkvT[(dch - 128) * 128 + cin] = v;
  }
  if (i < 16384) g_ctx[i] = 0.f;
  if (i < 512)   g_zk[i]  = 0.f;
}

// ---------------------------------------------------------------------------
// Phase A
// ---------------------------------------------------------------------------
__global__ void __launch_bounds__(256)
phaseA_ctx(const float* __restrict__ x, const u16_t* __restrict__ WkvT,
           float* __restrict__ g_ctx, float* __restrict__ g_zk,
           u16_t* __restrict__ xbf, int emit_bf) {
  extern __shared__ char smem[];
  u16_t* Xbf = (u16_t*)smem;                       // 128 x ES bf16
  u16_t* E   = (u16_t*)(smem + 128 * ES * 2);      // 128 x ES bf16
  u16_t* V   = (u16_t*)(smem + 2 * 128 * ES * 2);  // 128 x ES bf16

  const int tid  = threadIdx.x;
  const int lane = tid & 31;
  const int w    = tid >> 5;            // wave 0..7
  const int l15  = lane & 15;
  const int lh   = (lane >> 4) & 1;

  const int tile0 = blockIdx.x * 4;     // 1024 tiles of 128 tokens, 256/batch
  const int b     = tile0 >> 8;

  // persistent B fragments: wave w owns k-channels [16w,16w+16) and
  // v-channels [16w,16w+16) (rows 128+16w of Wkv^T). 8 frags = 64 VGPRs.
  v16bf bk0 = frag_bf16(WkvT, w * 16, 128, 0, lane);
  v16bf bk1 = frag_bf16(WkvT, w * 16, 128, 1, lane);
  v16bf bk2 = frag_bf16(WkvT, w * 16, 128, 2, lane);
  v16bf bk3 = frag_bf16(WkvT, w * 16, 128, 3, lane);
  v16bf bv0 = frag_bf16(WkvT, 128 + w * 16, 128, 0, lane);
  v16bf bv1 = frag_bf16(WkvT, 128 + w * 16, 128, 1, lane);
  v16bf bv2 = frag_bf16(WkvT, 128 + w * 16, 128, 2, lane);
  v16bf bv3 = frag_bf16(WkvT, 128 + w * 16, 128, 3, lane);

  // persistent context accumulators (wave -> two (h,mi,ni) 16x16 tiles of 16)
  const int tA = 2 * w, tB = 2 * w + 1;
  const int hA = tA >> 2, mA = (tA >> 1) & 1, nA = tA & 1;
  const int hB = tB >> 2, mB = (tB >> 1) & 1, nB = tB & 1;
  v8f ctx0 = {}; v8f ctx1 = {};
  float zk = 0.f;

  for (int t = 0; t < 4; ++t) {
    const long token0 = (long)(tile0 + t) * 128;

    // stage X tile: f32 -> bf16 LDS [token][128] (row stride ES)
    for (int i = tid; i < 4096; i += 256) {
      int row = i >> 5, c4 = i & 31;
      float4 vv = *(const float4*)(x + (token0 + row) * 128 + c4 * 4);
      *(uint2*)(Xbf + row * ES + c4 * 4) = cvt4(vv);
    }
    __syncthreads();

    // emit flat bf16 image of this tile (whole padded block) for phase B
    if (emit_bf) {
      u16_t* gtile = xbf + (size_t)(tile0 + t) * TILE_ELEMS;
      for (int i = tid; i < TILE_ELEMS / 8; i += 256) {   // 2176 x 16B chunks
        async_store_b128(gtile + i * 8, lds_off_of(Xbf + i * 8));
      }
    }

    // KV = X @ Wkv^T : loop M-tiles, B frags resident in registers
#pragma unroll
    for (int mt = 0; mt < 8; ++mt) {
      v16bf a0 = frag_bf16(Xbf, mt * 16, ES, 0, lane);
      v16bf a1 = frag_bf16(Xbf, mt * 16, ES, 1, lane);
      v16bf a2 = frag_bf16(Xbf, mt * 16, ES, 2, lane);
      v16bf a3 = frag_bf16(Xbf, mt * 16, ES, 3, lane);
      v8f ak = {}; v8f av = {};
      ak = wmma_bf16(a0, bk0, ak); av = wmma_bf16(a0, bv0, av);
      ak = wmma_bf16(a1, bk1, ak); av = wmma_bf16(a1, bv1, av);
      ak = wmma_bf16(a2, bk2, ak); av = wmma_bf16(a2, bv2, av);
      ak = wmma_bf16(a3, bk3, ak); av = wmma_bf16(a3, bv3, av);
      // D-tile: lane holds channel w*16+l15, tokens mt*16+lh*8+r (contiguous)
      FragU pe, pv;
#pragma unroll
      for (int r = 0; r < 8; ++r) {
        pe.s[r] = f2bf(__expf(ak[r]));
        pv.s[r] = f2bf(av[r]);
      }
      int och = w * 16 + l15;
      int toff = mt * 16 + lh * 8;
      *(uint4*)(E + och * ES + toff) = pe.u[0];   // one b128, conflict-free
      *(uint4*)(V + och * ES + toff) = pv.u[0];
    }
    __syncthreads();

    // Z partial sums: thread ch < 128 sums its E row over 128 tokens
    if (tid < 128) {
      const __bf16* er = (const __bf16*)(E + tid * ES);
      float s = 0.f;
#pragma unroll 8
      for (int n = 0; n < 128; ++n) s += (float)er[n];
      zk += s;
    }

    // context += E(32d x 128n) @ V^T(128n x 32c) per head; wave does 2 tiles
#pragma unroll
    for (int ks = 0; ks < 4; ++ks) {
      ctx0 = wmma_bf16(frag_bf16(E, hA * 32 + mA * 16, ES, ks, lane),
                       frag_bf16(V, hA * 32 + nA * 16, ES, ks, lane), ctx0);
      ctx1 = wmma_bf16(frag_bf16(E, hB * 32 + mB * 16, ES, ks, lane),
                       frag_bf16(V, hB * 32 + nB * 16, ES, ks, lane), ctx1);
    }
    if (emit_bf) wait_async0();   // async copies read Xbf; finish before overwrite
    __syncthreads();
  }

  // flush partials with fp32 atomics into g_ctx[b][h][d][e]
  {
    float* baseA = g_ctx + (b * 4 + hA) * 32 * 32;
    float* baseB = g_ctx + (b * 4 + hB) * 32 * 32;
#pragma unroll
    for (int r = 0; r < 8; ++r) {
      int dA = mA * 16 + r + lh * 8, eA = nA * 16 + l15;
      int dB = mB * 16 + r + lh * 8, eB = nB * 16 + l15;
      atomicAdd(baseA + dA * 32 + eA, ctx0[r]);
      atomicAdd(baseB + dB * 32 + eB, ctx1[r]);
    }
  }
  if (tid < 128) atomicAdd(g_zk + b * 128 + tid, zk);
}

// ---------------------------------------------------------------------------
// mid: M^T[b][c][k=h*32+d] = sum_e (ctx[b][h][d][e]/Z[b][k]) * w_out[h*32+e][c]
// ---------------------------------------------------------------------------
__global__ void __launch_bounds__(256)
mid_kernel(const float* __restrict__ g_ctx, const float* __restrict__ g_zk,
           const float* __restrict__ w_out, u16_t* __restrict__ Mt) {
  int i = blockIdx.x * 256 + threadIdx.x;     // 65536 items
  if (i >= 65536) return;
  int b   = i >> 14;
  int rem = i & 16383;
  int c   = rem & 127;
  int k   = rem >> 7;                         // h*32 + d
  int h = k >> 5, d = k & 31;
  const float* ctx = g_ctx + ((b * 4 + h) * 32 + d) * 32;
  float inv = 1.f / g_zk[b * 128 + k];
  float s = 0.f;
#pragma unroll
  for (int e = 0; e < 32; ++e) s += ctx[e] * w_out[(h * 32 + e) * 128 + c];
  Mt[((size_t)(b * 128 + c)) * 128 + k] = f2bf(s * inv);
}

// ---------------------------------------------------------------------------
// Phase B
// ---------------------------------------------------------------------------
__global__ void __launch_bounds__(256)
phaseB_out(const float* __restrict__ x, const u16_t* __restrict__ xbf, int use_bf,
           const u16_t* __restrict__ WqT, const u16_t* __restrict__ Mt,
           const float* __restrict__ b_out, const float* __restrict__ ln_s,
           const float* __restrict__ ln_b, float* __restrict__ out) {
  extern __shared__ char smem[];
  u16_t* Xbf  = (u16_t*)smem;                        // 128 x ES bf16, then S
  u16_t* S    = (u16_t*)smem;
  float* regB = (float*)(smem + 128 * ES * 2);       // Q then Y, 128 x XS f32
  float* muA  = (float*)(smem + 128 * ES * 2 + 128 * XS * 4);   // 128 f32
  float* rsA  = muA + 128;

  const int tid  = threadIdx.x;
  const int lane = tid & 31;
  const int w    = tid >> 5;
  const int l15  = lane & 15;
  const int lh   = (lane >> 4) & 1;

  const long token0 = (long)blockIdx.x * 128;
  const int  b      = (int)(token0 >> 15);
  const u16_t* Mtb  = Mt + (size_t)b * 128 * 128;

  // kick off X tile staging: one TDM descriptor (flat 34816 B copy) if available
  if (use_bf) {
    const u16_t* gtile = xbf + (size_t)blockIdx.x * TILE_ELEMS;
#if HAS_TDM
    if (w == 0) {
      v4u g0; v8i g1;
      tdm_desc(g0, g1, lds_off_of(Xbf), gtile);
      tdm_load(g0, g1);
    }
#else
    for (int i = tid; i < TILE_ELEMS / 8; i += 256)
      async_load_b128(lds_off_of(Xbf + i * 8), gtile + i * 8);
#endif
  } else {
    for (int i = tid; i < 4096; i += 256) {
      int row = i >> 5, c4 = i & 31;
      float4 vv = *(const float4*)(x + (token0 + row) * 128 + c4 * 4);
      *(uint2*)(Xbf + row * ES + c4 * 4) = cvt4(vv);
    }
  }

  // persistent B fragments while the DMA is in flight (64 VGPRs)
  v16bf bq0 = frag_bf16(WqT, w * 16, 128, 0, lane);
  v16bf bq1 = frag_bf16(WqT, w * 16, 128, 1, lane);
  v16bf bq2 = frag_bf16(WqT, w * 16, 128, 2, lane);
  v16bf bq3 = frag_bf16(WqT, w * 16, 128, 3, lane);
  v16bf bm0 = frag_bf16(Mtb, w * 16, 128, 0, lane);
  v16bf bm1 = frag_bf16(Mtb, w * 16, 128, 1, lane);
  v16bf bm2 = frag_bf16(Mtb, w * 16, 128, 2, lane);
  v16bf bm3 = frag_bf16(Mtb, w * 16, 128, 3, lane);

  if (use_bf) {
#if HAS_TDM
    if (w == 0) wait_tensor0();
#else
    wait_async0();
#endif
  }
  __syncthreads();

  // Q = X @ Wq^T -> regB [token][ch] f32 (wave owns 16 output channels)
#pragma unroll
  for (int mt = 0; mt < 8; ++mt) {
    v16bf a0 = frag_bf16(Xbf, mt * 16, ES, 0, lane);
    v16bf a1 = frag_bf16(Xbf, mt * 16, ES, 1, lane);
    v16bf a2 = frag_bf16(Xbf, mt * 16, ES, 2, lane);
    v16bf a3 = frag_bf16(Xbf, mt * 16, ES, 3, lane);
    v8f acc = {};
    acc = wmma_bf16(a0, bq0, acc);
    acc = wmma_bf16(a1, bq1, acc);
    acc = wmma_bf16(a2, bq2, acc);
    acc = wmma_bf16(a3, bq3, acc);
    int ch = w * 16 + l15;
#pragma unroll
    for (int r = 0; r < 8; ++r) regB[(mt * 16 + r + lh * 8) * XS + ch] = acc[r];
  }
  __syncthreads();

  // per-(token, head) softmax over 32 channels, *scale, bf16 into S (over Xbf)
  const float scale = 0.17677669529663689f;  // DIM_HEAD^-0.5
#pragma unroll
  for (int it = 0; it < 2; ++it) {
    int item = it * 256 + tid;               // 512 = 128 tokens * 4 heads
    int tok = item >> 2, h = item & 3;
    const float* qr = regB + tok * XS + h * 32;
    float mx = -3.0e38f;
#pragma unroll
    for (int j = 0; j < 32; ++j) mx = fmaxf(mx, qr[j]);
    float ev[32]; float sum = 0.f;
#pragma unroll
    for (int j = 0; j < 32; ++j) { ev[j] = __expf(qr[j] - mx); sum += ev[j]; }
    float inv = scale / sum;
    u16_t* sr = S + tok * ES + h * 32;
#pragma unroll
    for (int j = 0; j < 32; ++j) sr[j] = f2bf(ev[j] * inv);
  }
  __syncthreads();

  // Y = S @ M^T -> regB (Q dead)
#pragma unroll
  for (int mt = 0; mt < 8; ++mt) {
    v16bf a0 = frag_bf16(S, mt * 16, ES, 0, lane);
    v16bf a1 = frag_bf16(S, mt * 16, ES, 1, lane);
    v16bf a2 = frag_bf16(S, mt * 16, ES, 2, lane);
    v16bf a3 = frag_bf16(S, mt * 16, ES, 3, lane);
    v8f acc = {};
    acc = wmma_bf16(a0, bm0, acc);
    acc = wmma_bf16(a1, bm1, acc);
    acc = wmma_bf16(a2, bm2, acc);
    acc = wmma_bf16(a3, bm3, acc);
    int ch = w * 16 + l15;
#pragma unroll
    for (int r = 0; r < 8; ++r) regB[(mt * 16 + r + lh * 8) * XS + ch] = acc[r];
  }
  __syncthreads();

  // LayerNorm over channels (b_out folded in)
  if (tid < 128) {
    const float* yr = regB + tid * XS;
    float s = 0.f, s2 = 0.f;
#pragma unroll 8
    for (int c = 0; c < 128; ++c) { float v = yr[c] + b_out[c]; s += v; s2 += v * v; }
    float mu  = s * (1.f / 128.f);
    float var = s2 * (1.f / 128.f) - mu * mu;
    muA[tid] = mu;
    rsA[tid] = rsqrtf(var + 1e-6f);
  }
  __syncthreads();
  for (int i = tid; i < 16384; i += 256) {
    int tok = i >> 7, c = i & 127;
    float v = regB[tok * XS + c] + b_out[c];
    out[token0 * 128 + i] = (v - muA[tok]) * rsA[tok] * ln_s[c] + ln_b[c];
  }
}

// ---------------------------------------------------------------------------
extern "C" void kernel_launch(void* const* d_in, const int* in_sizes, int n_in,
                              void* d_out, int out_size, void* d_ws, size_t ws_size,
                              hipStream_t stream) {
  const float* x     = (const float*)d_in[0];
  const float* w_qkv = (const float*)d_in[1];
  const float* w_out = (const float*)d_in[2];
  const float* b_out = (const float*)d_in[3];
  const float* ln_s  = (const float*)d_in[4];
  const float* ln_b  = (const float*)d_in[5];
  float* out = (float*)d_out;

  char* ws = (char*)d_ws;
  float* g_ctx = (float*)(ws + 0);        // 4*4*32*32 f32  = 65536 B
  float* g_zk  = (float*)(ws + 65536);    // 4*128 f32      = 2048 B
  u16_t* WqT   = (u16_t*)(ws + 67584);    // 128*128 bf16   = 32768 B
  u16_t* WkvT  = (u16_t*)(ws + 100352);   // 256*128 bf16   = 65536 B
  u16_t* Mt    = (u16_t*)(ws + 165888);   // 4*128*128 bf16 = 131072 B
  constexpr size_t WS_FIXED = 296960;
  u16_t* xbf   = (u16_t*)(ws + WS_FIXED); // 1024 tiles x 34816 B (~34 MiB, optional)
  const size_t need = WS_FIXED + (size_t)1024 * TILE_ELEMS * 2;
  const int use_bf = (ws_size >= need) ? 1 : 0;

  prep_kernel<<<192, 256, 0, stream>>>(w_qkv, WqT, WkvT, g_ctx, g_zk);
  phaseA_ctx<<<256, 256, SMEM_A, stream>>>(x, WkvT, g_ctx, g_zk, xbf, use_bf);
  mid_kernel<<<256, 256, 0, stream>>>(g_ctx, g_zk, w_out, Mt);
  phaseB_out<<<1024, 256, SMEM_B, stream>>>(x, xbf, use_bf, WqT, Mt, b_out,
                                            ln_s, ln_b, out);
}
